// AdditiveAttention_33500744909309
// MI455X (gfx1250) — compile-verified, matrix operand order checked by reading, not run on hardware
//
#include <hip/hip_runtime.h>

// Problem constants (match reference)
#define BN 4
#define QQ 512
#define KK 512
#define DQ 256
#define DV 256
#define HH 256
#define NEGV (-1e8f)

typedef __attribute__((ext_vector_type(16))) __bf16 v16bf;
typedef __attribute__((ext_vector_type(8)))  __bf16 v8bf;
typedef __attribute__((ext_vector_type(8)))  float  v8f;

// Native gfx1250 tanh (V_TANH_F32) instead of libm software sequence.
#if __has_builtin(__builtin_amdgcn_tanhf)
__device__ __forceinline__ float hw_tanh(float x) { return __builtin_amdgcn_tanhf(x); }
#else
__device__ __forceinline__ float hw_tanh(float x) {
    float r;
    asm volatile("v_tanh_f32 %0, %1" : "=v"(r) : "v"(x));
    return r;
}
#endif

// ---------------------------------------------------------------------------
// Elementwise f32 -> bf16 conversion, 8 elements per thread (b128 in/out).
// ---------------------------------------------------------------------------
__global__ void cvt_bf16_kernel(const float* __restrict__ src,
                                __bf16* __restrict__ dst, int n8) {
    const int i = blockIdx.x * blockDim.x + threadIdx.x;
    if (i >= n8) return;
    const float4* s = (const float4*)src + (size_t)i * 2;
    float4 f0 = s[0], f1 = s[1];
    v8bf o;
    o[0] = (__bf16)f0.x; o[1] = (__bf16)f0.y; o[2] = (__bf16)f0.z; o[3] = (__bf16)f0.w;
    o[4] = (__bf16)f1.x; o[5] = (__bf16)f1.y; o[6] = (__bf16)f1.z; o[7] = (__bf16)f1.w;
    ((v8bf*)dst)[i] = o;
}

// ---------------------------------------------------------------------------
// Batched transpose + convert: values (b, K, DV) f32 -> valuesT (b, DV, K) bf16
// LDS-tiled 32x32 so both global read and write are coalesced.
// ---------------------------------------------------------------------------
__global__ void transpose_cvt_kernel(const float* __restrict__ V,
                                     __bf16* __restrict__ VT) {
    __shared__ __bf16 tile[32][33];
    const int b   = blockIdx.y;
    const int ntn = DV / 32;                       // tiles along n
    const int k0  = (blockIdx.x / ntn) * 32;
    const int n0  = (blockIdx.x % ntn) * 32;
    const float*  src = V  + (size_t)b * KK * DV;
    __bf16*       dst = VT + (size_t)b * DV * KK;
    const int tx = threadIdx.x, ty = threadIdx.y;
#pragma unroll
    for (int j = 0; j < 32; j += 8)
        tile[ty + j][tx] = (__bf16)src[(size_t)(k0 + ty + j) * DV + n0 + tx];
    __syncthreads();
#pragma unroll
    for (int j = 0; j < 32; j += 8)
        dst[(size_t)(n0 + ty + j) * KK + k0 + tx] = tile[tx][ty + j];
}

// ---------------------------------------------------------------------------
// Generic batched bf16 GEMM: C[m][n] = sum_k A[m][k] * Bt[n][k]
//   A : M x Kd bf16 row-major  (batch stride sA elements)
//   Bt: Hn x Kd bf16 row-major (batch stride sB elements)  == B^T
//   C : M x Hn f32 row-major   (batch stride sC elements)
// One 16x64 output tile per wave (4 accumulators share one A fragment):
// 4 v_wmma_f32_16x16x32_bf16 per 32-k step vs 10 b128 loads.
// ---------------------------------------------------------------------------
__device__ __forceinline__ v16bf load_frag(const __bf16* __restrict__ base,
                                           int k1, int k2) {
    v8bf lo = *(const v8bf*)(base + k1);   // 16B
    v8bf hi = *(const v8bf*)(base + k2);   // 16B
    v16bf r;
#pragma unroll
    for (int i = 0; i < 8; ++i) { r[i] = lo[i]; r[8 + i] = hi[i]; }
    return r;
}

__global__ void gemm_bf16_wmma_kernel(const __bf16* __restrict__ A,
                                      const __bf16* __restrict__ Bt,
                                      float* __restrict__ C,
                                      int M, int Kd, int Hn,
                                      long sA, long sB, long sC) {
    const int b = blockIdx.y;
    A  += (long)b * sA;
    Bt += (long)b * sB;
    C  += (long)b * sC;

    const int wave = blockIdx.x * 8 + (threadIdx.x >> 5);
    const int lane = threadIdx.x & 31;
    const int ncg  = Hn >> 6;                      // 64-wide col groups
    const int m0   = (wave / ncg) << 4;
    const int n0   = (wave % ncg) << 6;
    const int half = lane >> 4;
    const int l15  = lane & 15;

    const __bf16* __restrict__ arow  = A  + (long)(m0 + l15) * Kd;
    const __bf16* __restrict__ brow0 = Bt + (long)(n0 + l15) * Kd;
    const __bf16* __restrict__ brow1 = brow0 + (long)16 * Kd;
    const __bf16* __restrict__ brow2 = brow0 + (long)32 * Kd;
    const __bf16* __restrict__ brow3 = brow0 + (long)48 * Kd;

    v8f acc0 = {}, acc1 = {}, acc2 = {}, acc3 = {};
    for (int kk = 0; kk < Kd; kk += 32) {
        const int k1 = kk + half * 8;
        const int k2 = k1 + 16;
        __builtin_prefetch(arow + kk + 64, 0, 1);  // global_prefetch_b8
        v16bf af = load_frag(arow,  k1, k2);
        v16bf b0 = load_frag(brow0, k1, k2);
        v16bf b1 = load_frag(brow1, k1, k2);
        v16bf b2 = load_frag(brow2, k1, k2);
        v16bf b3 = load_frag(brow3, k1, k2);
        acc0 = __builtin_amdgcn_wmma_f32_16x16x32_bf16(false, af, false, b0, (short)0, acc0, false, false);
        acc1 = __builtin_amdgcn_wmma_f32_16x16x32_bf16(false, af, false, b1, (short)0, acc1, false, false);
        acc2 = __builtin_amdgcn_wmma_f32_16x16x32_bf16(false, af, false, b2, (short)0, acc2, false, false);
        acc3 = __builtin_amdgcn_wmma_f32_16x16x32_bf16(false, af, false, b3, (short)0, acc3, false, false);
    }

#pragma unroll
    for (int r = 0; r < 8; ++r) {
        const long row = m0 + r + half * 8;
        float* cp = C + row * (long)Hn + n0 + l15;
        cp[0]  = acc0[r];
        cp[16] = acc1[r];
        cp[32] = acc2[r];
        cp[48] = acc3[r];
    }
}

// ---------------------------------------------------------------------------
// Fused score + mask + softmax. One block per (b,q), 512 threads = 1 key each.
// Inner loop uses the native V_TANH_F32 transcendental.
// Writes f32 weights to d_out and a bf16 shadow copy for the AV WMMA GEMM.
// ---------------------------------------------------------------------------
__global__ void score_softmax_kernel(const float* __restrict__ qf,
                                     const float* __restrict__ kf,
                                     const float* __restrict__ wv,
                                     const unsigned char* __restrict__ mask,
                                     float* __restrict__ weights,
                                     __bf16* __restrict__ weights_bf) {
    __shared__ float sq[HH];
    __shared__ float sw[HH];
    __shared__ float red[KK];

    const int b = blockIdx.x / QQ;
    const int q = blockIdx.x % QQ;
    const int t = threadIdx.x;                // key index 0..511

    if (t < HH) {
        sq[t] = qf[((size_t)b * QQ + q) * HH + t];
        sw[t] = wv[t];
    }
    __syncthreads();

    const float* __restrict__ krow = kf + ((size_t)b * KK + t) * HH;
    float s = 0.f;
#pragma unroll 4
    for (int h = 0; h < HH; h += 4) {
        float4 kv = *(const float4*)(krow + h);
        s = fmaf(sw[h],     hw_tanh(sq[h]     + kv.x), s);
        s = fmaf(sw[h + 1], hw_tanh(sq[h + 1] + kv.y), s);
        s = fmaf(sw[h + 2], hw_tanh(sq[h + 2] + kv.z), s);
        s = fmaf(sw[h + 3], hw_tanh(sq[h + 3] + kv.w), s);
    }
    if (mask[((size_t)b * QQ + q) * KK + t]) s = NEGV;

    red[t] = s;
    __syncthreads();
    for (int off = 256; off > 0; off >>= 1) {
        if (t < off) red[t] = fmaxf(red[t], red[t + off]);
        __syncthreads();
    }
    const float mx = red[0];
    __syncthreads();
    const float e = __expf(s - mx);
    red[t] = e;
    __syncthreads();
    for (int off = 256; off > 0; off >>= 1) {
        if (t < off) red[t] += red[t + off];
        __syncthreads();
    }
    const float w = e * (1.f / red[0]);
    const size_t idx = ((size_t)b * QQ + q) * KK + t;
    weights[idx]    = w;
    weights_bf[idx] = (__bf16)w;
}

// ---------------------------------------------------------------------------
extern "C" void kernel_launch(void* const* d_in, const int* in_sizes, int n_in,
                              void* d_out, int out_size, void* d_ws, size_t ws_size,
                              hipStream_t stream) {
    const float*         queries = (const float*)d_in[0];          // (4,512,256)
    const float*         keys    = (const float*)d_in[1];          // (4,512,256)
    const float*         values  = (const float*)d_in[2];          // (4,512,256)
    const unsigned char* mask    = (const unsigned char*)d_in[3];  // (4,512,512) bool
    const float*         Wq      = (const float*)d_in[4];          // (256,256)
    const float*         Wk      = (const float*)d_in[5];          // (256,256)
    const float*         wv      = (const float*)d_in[6];          // (256,)

    float* attn    = (float*)d_out;                                // (4,512,256)
    float* weights = (float*)d_out + (size_t)BN * QQ * DV;         // (4,512,512)

    // Workspace layout (7.25 MB peak):
    char* w = (char*)d_ws;
    float*  qf   = (float*)(w);                                    // 2 MB
    float*  kfp  = (float*)(w + (2u << 20));                       // 2 MB
    __bf16* qb   = (__bf16*)(w + (4u << 20));                      // 1 MB
    __bf16* kb   = (__bf16*)(w + (5u << 20));                      // 1 MB
    __bf16* wb16 = (__bf16*)(w + (4u << 20));                      // 2 MB (reuses qb+kb after proj)
    __bf16* Wqb  = (__bf16*)(w + (6u << 20));                      // 128 KB
    __bf16* Wkb  = (__bf16*)(w + (6u << 20) + (128u << 10));       // 128 KB
    __bf16* vT   = (__bf16*)(w + (6u << 20) + (256u << 10));       // 1 MB

    // 1) f32 -> bf16 conversions
    const int nQ8 = BN * QQ * DQ / 8;         // 65536
    const int nW8 = HH * DQ / 8;              // 8192
    cvt_bf16_kernel<<<(nQ8 + 255) / 256, 256, 0, stream>>>(queries, qb, nQ8);
    cvt_bf16_kernel<<<(nQ8 + 255) / 256, 256, 0, stream>>>(keys,    kb, nQ8);
    cvt_bf16_kernel<<<(nW8 + 255) / 256, 256, 0, stream>>>(Wq,      Wqb, nW8);
    cvt_bf16_kernel<<<(nW8 + 255) / 256, 256, 0, stream>>>(Wk,      Wkb, nW8);

    // 2) values (b,K,DV) f32 -> valuesT (b,DV,K) bf16
    {
        dim3 g((KK / 32) * (DV / 32), BN);
        dim3 blk(32, 8);
        transpose_cvt_kernel<<<g, blk, 0, stream>>>(values, vT);
    }

    // 3) Projections: qf = qb * Wqb^T, kf = kb * Wkb^T  (M=2048, Kd=256, Hn=256)
    {
        const int M = BN * QQ;
        const int waves = (M / 16) * (HH / 64);   // 512
        dim3 g(waves / 8, 1);
        gemm_bf16_wmma_kernel<<<g, 256, 0, stream>>>(qb, Wqb, qf,  M, DQ, HH, 0, 0, 0);
        gemm_bf16_wmma_kernel<<<g, 256, 0, stream>>>(kb, Wkb, kfp, M, DQ, HH, 0, 0, 0);
    }

    // 4) Fused score + softmax (also emits bf16 weights into ws)
    score_softmax_kernel<<<BN * QQ, KK, 0, stream>>>(qf, kfp, wv, mask, weights, wb16);

    // 5) AV GEMM per batch: attn[b] = weights[b] (512x512) * values[b] (512x256)
    {
        const int waves = (QQ / 16) * (DV / 64);  // 128 per batch
        dim3 g(waves / 8, BN);
        gemm_bf16_wmma_kernel<<<g, 256, 0, stream>>>(
            wb16, vT, attn, QQ, KK, DV,
            (long)QQ * KK, (long)DV * KK, (long)QQ * DV);
    }
}
